// att_conv_normalize_free_472446402607
// MI455X (gfx1250) — matrix operand order repaired and checked by reading, not compiled, not run on hardware
//
#include <hip/hip_runtime.h>

typedef __attribute__((ext_vector_type(16))) _Float16 v16h;
typedef __attribute__((ext_vector_type(8)))  _Float16 v8h;
typedef __attribute__((ext_vector_type(8)))  float    v8f;

#define HW_  65536   // H*W = 256*256
#define CIN_ 64

static __device__ __forceinline__ v8f wmma_f16(v16h a, v16h b, v8f c) {
  // D = A(16x32 f16) * B(32x16 f16) + C(16x16 f32)
  return __builtin_amdgcn_wmma_f32_16x16x32_f16(false, a, false, b, (short)0, c,
                                                false, false);
}

__global__ __launch_bounds__(256) void att_fused_wmma_472446402607(
    const float* __restrict__ kv, const float* __restrict__ qin,
    const float* __restrict__ key_w, const float* __restrict__ key_b,
    const float* __restrict__ value_w, const float* __restrict__ value_b,
    const float* __restrict__ query_w, const float* __restrict__ query_b,
    const float* __restrict__ scale_p,
    const float* __restrict__ d1_w, const float* __restrict__ d2_w,
    const float* __restrict__ d3_w, const float* __restrict__ d3_b,
    float* __restrict__ out, int num_tiles)
{
  const int lane  = threadIdx.x & 31;
  const int laneN = lane & 15;   // pixel (N) / row (M) index inside fragment
  const int half  = lane >> 4;   // lane half
  const float scale = scale_p[0];

  // ---- A-fragment loader: 16x32 f16 tile of row-major [O x C] weight ----
  // A k order used here (contiguous, B matches): lane holds row M=lane%16; VGPR j, half h ->
  // k = 16*h + 2*j + {0,1}
  auto loadA = [&](const float* w, int ldc, int mbase, int mmax,
                   int kbase, int kmax) -> v16h {
    v16h a;
#pragma unroll
    for (int j = 0; j < 8; ++j) {
      const int kl = (half << 4) + (j << 1);
#pragma unroll
      for (int e = 0; e < 2; ++e) {
        float v = 0.0f;
        if (laneN < mmax && (kl + e) < kmax)
          v = w[(mbase + laneN) * ldc + kbase + kl + e];
        a[2 * j + e] = (_Float16)v;
      }
    }
    return a;
  };

  // Weight fragments (kept resident; amortized over the tile loop)
  v16h aK00 = loadA(key_w,   64,  0, 16,  0, 32), aK01 = loadA(key_w,   64,  0, 16, 32, 32);
  v16h aK10 = loadA(key_w,   64, 16, 16,  0, 32), aK11 = loadA(key_w,   64, 16, 16, 32, 32);
  v16h aV00 = loadA(value_w, 64,  0, 16,  0, 32), aV01 = loadA(value_w, 64,  0, 16, 32, 32);
  v16h aV10 = loadA(value_w, 64, 16, 16,  0, 32), aV11 = loadA(value_w, 64, 16, 16, 32, 32);
  v16h aQ00 = loadA(query_w, 64,  0, 16,  0, 32), aQ01 = loadA(query_w, 64,  0, 16, 32, 32);
  v16h aQ10 = loadA(query_w, 64, 16, 16,  0, 32), aQ11 = loadA(query_w, 64, 16, 16, 32, 32);
  v16h aD1  = loadA(d1_w, 32, 0, 16, 0, 32);   // 16x32, exact
  v16h aD2  = loadA(d2_w, 16, 0,  8, 0, 16);   // 8x16 zero-padded to 16x32
  v16h aD3  = loadA(d3_w,  8, 0, 16, 0,  8);   // 16x8 zero-padded to 16x32

  // Biases kept packed f16 (halves register cost); D-frag lane: M = 8*half + r
  auto loadBiasPk = [&](const float* b, int off) -> v8h {
    v8h r;
#pragma unroll
    for (int i = 0; i < 8; ++i) r[i] = (_Float16)b[off + 8 * half + i];
    return r;
  };
  v8h pK0 = loadBiasPk(key_b, 0),   pK1 = loadBiasPk(key_b, 16);
  v8h pV0 = loadBiasPk(value_b, 0), pV1 = loadBiasPk(value_b, 16);
  v8h pQ0 = loadBiasPk(query_b, 0), pQ1 = loadBiasPk(query_b, 16);
  v8h pO  = loadBiasPk(d3_b, 0);

  auto unpack = [&](v8h p) -> v8f {
    v8f c;
#pragma unroll
    for (int i = 0; i < 8; ++i) c[i] = (float)p[i];
    return c;
  };

  // ---- B-fragment loader: 32 channels x 16 pixels, f32 -> f16 ----
  // B k order used here (contiguous, matches A): lane holds column N=lane%16; half h covers
  // k = 16*h + 2j + {0,1}. Each dword load: lanes 0-15 and 16-31 each hit a
  // contiguous 64B pixel run.
  auto loadB = [&](const float* in, int base, int kbase) -> v16h {
    v16h r;
    const int c0 = kbase + (half << 4);
#pragma unroll
    for (int j = 0; j < 8; ++j) {
      float x0 = in[base + (c0 + 2 * j) * HW_];
      float x1 = in[base + (c0 + 2 * j + 1) * HW_];
      r[2 * j]     = (_Float16)x0;
      r[2 * j + 1] = (_Float16)x1;
    }
    return r;
  };

  const int wid    = blockIdx.x * (blockDim.x >> 5) + (threadIdx.x >> 5);
  const int nwaves = gridDim.x * (blockDim.x >> 5);

#pragma unroll 1
  for (int t = wid; t < num_tiles; t += nwaves) {
    const int p0   = t << 4;                 // first flat pixel of tile
    const int b    = p0 >> 16;               // / HW_
    const int hw   = (p0 & (HW_ - 1)) + laneN;
    const int base = b * (CIN_ * HW_) + hw;  // channel-0 element offset

    v16h bKV0 = loadB(kv,  base, 0);
    v16h bKV1 = loadB(kv,  base, 32);
    v16h bQ0  = loadB(qin, base, 0);
    v16h bQ1  = loadB(qin, base, 32);

    // k and q first (8 WMMAs), so qf/bQ die before vf comes alive
    v8f kf0 = unpack(pK0), kf1 = unpack(pK1);
    v8f qf0 = unpack(pQ0), qf1 = unpack(pQ1);
    kf0 = wmma_f16(aK00, bKV0, kf0); kf0 = wmma_f16(aK01, bKV1, kf0);
    kf1 = wmma_f16(aK10, bKV0, kf1); kf1 = wmma_f16(aK11, bKV1, kf1);
    qf0 = wmma_f16(aQ00, bQ0,  qf0); qf0 = wmma_f16(aQ01, bQ1,  qf0);
    qf1 = wmma_f16(aQ10, bQ0,  qf1); qf1 = wmma_f16(aQ11, bQ1,  qf1);

    // att = sigmoid(k*q*scale); kf/qf are consumed here
    v8f att0, att1;
#pragma unroll
    for (int r = 0; r < 8; ++r) {
      float t0 = kf0[r] * qf0[r] * scale;
      float t1 = kf1[r] * qf1[r] * scale;
      att0[r] = 1.0f / (1.0f + __expf(-t0));
      att1[r] = 1.0f / (1.0f + __expf(-t1));
    }

    // value conv (4 WMMAs) and gate
    v8f vf0 = unpack(pV0), vf1 = unpack(pV1);
    vf0 = wmma_f16(aV00, bKV0, vf0); vf0 = wmma_f16(aV01, bKV1, vf0);
    vf1 = wmma_f16(aV10, bKV0, vf1); vf1 = wmma_f16(aV11, bKV1, vf1);

    v8f x0, x1;
#pragma unroll
    for (int r = 0; r < 8; ++r) {
      x0[r] = att0[r] * vf0[r];
      x1[r] = att1[r] * vf1[r];
    }

    // Re-layout x (32x16 D-frags) into a 32x16 B-fragment via lane-half swap
    v16h bx;
#pragma unroll
    for (int r = 0; r < 8; ++r) {
      float s0 = __shfl_xor(x0[r], 16, 32);
      float s1 = __shfl_xor(x1[r], 16, 32);
      bx[r]     = (_Float16)(half ? s1    : x0[r]);
      bx[8 + r] = (_Float16)(half ? x1[r] : s0);
    }

    // h1 = relu(d1_w @ x)   16x16
    v8f h1 = {};
    h1 = wmma_f16(aD1, bx, h1);
#pragma unroll
    for (int r = 0; r < 8; ++r) h1[r] = fmaxf(h1[r], 0.0f);

    // h1 -> B fragment, K padded 16->32 (upper half zero)
    v16h bh1;
#pragma unroll
    for (int r = 0; r < 8; ++r) {
      float s0 = __shfl_xor(h1[r], 16, 32);
      bh1[r]     = (_Float16)(half ? 0.0f : h1[r]);
      bh1[8 + r] = (_Float16)(half ? 0.0f : s0);
    }

    // h2 = relu(d2_w @ h1)  8x16 (rows 8..15 are zero by A padding)
    v8f h2 = {};
    h2 = wmma_f16(aD2, bh1, h2);
#pragma unroll
    for (int r = 0; r < 8; ++r) h2[r] = fmaxf(h2[r], 0.0f);

    // h2 -> B fragment, only k=0..7 matter (A padded)
    v16h bh2;
#pragma unroll
    for (int r = 0; r < 8; ++r) {
      bh2[r]     = (_Float16)(half ? 0.0f : h2[r]);
      bh2[8 + r] = (_Float16)0.0f;
    }

    // out = d3_w @ h2 + d3_b   16x16
    v8f of = unpack(pO);
    of = wmma_f16(aD3, bh2, of);

    const int obase = b * (16 * HW_) + hw;
#pragma unroll
    for (int r = 0; r < 8; ++r)
      out[obase + (8 * half + r) * HW_] = of[r];
  }
}

extern "C" void kernel_launch(void* const* d_in, const int* in_sizes, int n_in,
                              void* d_out, int out_size, void* d_ws, size_t ws_size,
                              hipStream_t stream) {
  (void)n_in; (void)out_size; (void)d_ws; (void)ws_size;
  const float* kv      = (const float*)d_in[0];
  const float* qin     = (const float*)d_in[1];
  const float* key_w   = (const float*)d_in[2];
  const float* key_b   = (const float*)d_in[3];
  const float* value_w = (const float*)d_in[4];
  const float* value_b = (const float*)d_in[5];
  const float* query_w = (const float*)d_in[6];
  const float* query_b = (const float*)d_in[7];
  const float* scale   = (const float*)d_in[8];
  const float* d1w     = (const float*)d_in[9];
  const float* d2w     = (const float*)d_in[10];
  const float* d3w     = (const float*)d_in[11];
  const float* d3b     = (const float*)d_in[12];
  float* out           = (float*)d_out;

  const int pixels    = in_sizes[0] / CIN_;   // B*H*W = 1,048,576
  const int num_tiles = pixels >> 4;          // 65,536 tiles of 16 pixels

  dim3 grid(1024), block(256);                // 8192 waves, 8 tiles/wave
  att_fused_wmma_472446402607<<<grid, block, 0, stream>>>(
      kv, qin, key_w, key_b, value_w, value_b, query_w, query_b, scale,
      d1w, d2w, d3w, d3b, out, num_tiles);
}
